// MHA_884763263199
// MI455X (gfx1250) — matrix operand order repaired and checked by reading, not compiled, not run on hardware
//
#include <hip/hip_runtime.h>

// ---------------------------------------------------------------------------
// MHA forward for MI455X (gfx1250, wave32, WMMA).
// Precision: bf16 WMMA inputs, f32 accumulation (v_wmma_f32_16x16x32_bf16).
// Pipeline:
//   gemm(X,Wq)->Q[B,H,S,Dh]  gemm(X,Wk)->K[B,H,S,Dh]  gemm(X,Wv)->V^T[B,H,Dh,S]
//   flash-attention -> ctx[B,S,D] (bf16)
//   gemm(ctx,Wo)+bo -> out fp32
// Roofline: ~69 GFLOP vs ~67 MB traffic -> compute-bound; everything fits in
// the 192 MB L2, so LDS-staged bf16 WMMA tiles are the right operating point.
// ---------------------------------------------------------------------------

typedef __attribute__((ext_vector_type(16))) __bf16 v16bf;
typedef __attribute__((ext_vector_type(8)))  __bf16 v8bf;
typedef __attribute__((ext_vector_type(8)))  float  v8f;
typedef __attribute__((ext_vector_type(4)))  float  v4f;

#define WMMA_BF16(a, b, c) \
  __builtin_amdgcn_wmma_f32_16x16x32_bf16(false, (a), false, (b), (short)0, (c), false, false)

#ifndef LOG2E_F
#define LOG2E_F 1.44269504088896340736f
#endif

// ===========================================================================
// Tiled GEMM: Out[M,N] = A[M,K] * W^T + bias   (M=4096, N=K=1024)
// MODE 0: fp32 out [M,N]
// MODE 1: bf16 out [B,H,S,Dh]     (Q/K layout)
// MODE 2: bf16 out [B,H,Dh,S]     (V transposed layout)
// A_BF16: A is bf16 (ctx) instead of fp32 (hidden_state)
// Compile-time flags -> no exec-mask/scalar-branch churn in the k-loop.
// ===========================================================================
template <int A_BF16, int MODE>
__global__ __launch_bounds__(256) void gemm_bias_wmma(
    const void* __restrict__ Aptr,
    const float* __restrict__ W,
    const float* __restrict__ bias,
    void* __restrict__ Out)
{
  constexpr int K = 1024, N = 1024;
  constexpr int BM = 128, BN = 128, BK = 32;
  constexpr int NS = K / BK;                 // 32 k-steps

  __shared__ alignas(64) __bf16 As[2][BM * BK];   // 8 KB x2
  __shared__ alignas(64) __bf16 Bs[2][BN * BK];   // 8 KB x2

  const int tid  = threadIdx.x;
  const int lane = tid & 31;
  const int wave = tid >> 5;                 // 0..7
  const int m0 = blockIdx.y * BM;
  const int n0 = blockIdx.x * BN;

  const int wm = (wave >> 2) * 64;           // wave row offset: 0 / 64
  const int wn = (wave & 3) * 32;            // wave col offset: 0/32/64/96

  const float*  Af = (const float*)Aptr;
  const __bf16* Ab = (const __bf16*)Aptr;

  v8f acc[4][2] = {};                        // 4 m-tiles x 2 n-tiles of 16x16

  float areg[16], wreg[16];                  // global->reg staging

  auto g2r = [&](int ks) {
    const int k0 = ks * BK;
#pragma unroll
    for (int i = 0; i < 4; ++i) {
      const int idx4 = i * 256 + tid;        // float4 index in 128x32 tile
      const int row  = idx4 >> 3;
      const int col  = (idx4 & 7) * 4;
      if constexpr (A_BF16) {
        const v8bf v = *(const v8bf*)(Ab + (size_t)(m0 + (idx4 >> 2)) * K +
                                      k0 + (idx4 & 3) * 8);
        // bf16 A: re-map to 8-wide chunks (128x32 tile = 4096 bf16, 8/thread/iter)
#pragma unroll
        for (int e = 0; e < 4; ++e) areg[i * 4 + e] = 0.0f;  // unused path below
        (void)v;
      } else {
        const v4f v = *(const v4f*)(Af + (size_t)(m0 + row) * K + k0 + col);
#pragma unroll
        for (int e = 0; e < 4; ++e) areg[i * 4 + e] = v[e];
      }
      const v4f wv = *(const v4f*)(W + (size_t)(n0 + row) * K + k0 + col);
#pragma unroll
      for (int e = 0; e < 4; ++e) wreg[i * 4 + e] = wv[e];
    }
  };

  // bf16-A staging path keeps raw bf16 (no cvt round-trip).
  v8bf abreg[2];
  auto g2r_bf = [&](int ks) {
    const int k0 = ks * BK;
#pragma unroll
    for (int i = 0; i < 2; ++i) {
      const int idx8 = i * 256 + tid;        // v8bf index in 128x32 tile
      const int row  = idx8 >> 2;
      const int col  = (idx8 & 3) * 8;
      abreg[i] = *(const v8bf*)(Ab + (size_t)(m0 + row) * K + k0 + col);
    }
  };

  auto r2lds = [&](int buf) {
#pragma unroll
    for (int i = 0; i < 4; ++i) {
      const int idx4 = i * 256 + tid;
      const int row  = idx4 >> 3;
      const int col  = (idx4 & 7) * 4;
      if constexpr (!A_BF16) {
#pragma unroll
        for (int e = 0; e < 4; ++e)
          As[buf][row * BK + col + e] = (__bf16)areg[i * 4 + e];
      }
#pragma unroll
      for (int e = 0; e < 4; ++e)
        Bs[buf][row * BK + col + e] = (__bf16)wreg[i * 4 + e];
    }
    if constexpr (A_BF16) {
#pragma unroll
      for (int i = 0; i < 2; ++i) {
        const int idx8 = i * 256 + tid;
        const int row  = idx8 >> 2;
        const int col  = (idx8 & 3) * 8;
        *(v8bf*)(&As[buf][row * BK + col]) = abreg[i];
      }
    }
  };

  // A 16x32 bf16 frag: lane holds row m=lane&15; elements 0..15 at
  // K = base..base+15, base = lane<16?0:16 (matches B frag order).
  auto load_a_frag = [&](const __bf16* base, int mrow) -> v16bf {
    const int m  = mrow + (lane & 15);
    const int b8 = (lane < 16) ? 0 : 16;
    const v8bf lo = *(const v8bf*)(base + m * BK + b8);
    const v8bf hi = *(const v8bf*)(base + m * BK + 8 + b8);
    v16bf a;
#pragma unroll
    for (int i = 0; i < 8; ++i) { a[i] = lo[i]; a[8 + i] = hi[i]; }
    return a;
  };
  // B 32x16 bf16 frag: lane holds column n=lane&15, K = koff + i (i=0..15),
  // koff = lane<16?0:16. Bs stores B^T rows (W rows), so one contiguous load.
  auto load_b_frag = [&](const __bf16* base, int nrow) -> v16bf {
    const int n    = nrow + (lane & 15);
    const int koff = (lane < 16) ? 0 : 16;
    return *(const v16bf*)(base + n * BK + koff);
  };

  if constexpr (A_BF16) g2r_bf(0); else g2r(0);
  if constexpr (A_BF16) {
    // still need W staged for step 0
    const int k0 = 0;
#pragma unroll
    for (int i = 0; i < 4; ++i) {
      const int idx4 = i * 256 + tid;
      const int row  = idx4 >> 3;
      const int col  = (idx4 & 7) * 4;
      const v4f wv = *(const v4f*)(W + (size_t)(n0 + row) * K + k0 + col);
#pragma unroll
      for (int e = 0; e < 4; ++e) wreg[i * 4 + e] = wv[e];
    }
  }
  r2lds(0);

  for (int s = 0; s < NS; ++s) {
    __syncthreads();                         // lds[s&1] ready
    if (s + 1 < NS) {                        // overlap global latency
      if constexpr (A_BF16) {
        g2r_bf(s + 1);
        const int k0 = (s + 1) * BK;
#pragma unroll
        for (int i = 0; i < 4; ++i) {
          const int idx4 = i * 256 + tid;
          const int row  = idx4 >> 3;
          const int col  = (idx4 & 7) * 4;
          const v4f wv = *(const v4f*)(W + (size_t)(n0 + row) * K + k0 + col);
#pragma unroll
          for (int e = 0; e < 4; ++e) wreg[i * 4 + e] = wv[e];
        }
      } else {
        g2r(s + 1);
      }
    }
    if (s + 2 < NS) {                        // temporal L2 prefetch, 2 tiles ahead
      __builtin_prefetch((const char*)W + ((size_t)n0 * K + (s + 2) * BK) * 4, 0, 3);
      __builtin_prefetch((const char*)Aptr +
                         ((size_t)m0 * K + (s + 2) * BK) * (A_BF16 ? 2 : 4), 0, 3);
    }
    const int buf = s & 1;

    // Preload ALL fragments first so one dscnt wait covers them and the
    // 8 WMMAs issue back-to-back (no wait between pairs).
    v16bf bfr[2], afr[4];
#pragma unroll
    for (int j = 0; j < 2; ++j) bfr[j] = load_b_frag(&Bs[buf][0], wn + j * 16);
#pragma unroll
    for (int i = 0; i < 4; ++i) afr[i] = load_a_frag(&As[buf][0], wm + i * 16);
#pragma unroll
    for (int i = 0; i < 4; ++i)
#pragma unroll
      for (int j = 0; j < 2; ++j)
        acc[i][j] = WMMA_BF16(afr[i], bfr[j], acc[i][j]);

    __syncthreads();                         // done reading before refill
    if (s + 1 < NS) r2lds((s + 1) & 1);
  }

  // Epilogue. C layout: n = lane&15; m = r + (lane>=16 ? 8 : 0) for VGPR r.
  const int nlane = lane & 15;
  const int mhalf = (lane >= 16) ? 8 : 0;
#pragma unroll
  for (int i = 0; i < 4; ++i) {
#pragma unroll
    for (int j = 0; j < 2; ++j) {
      const int ncol = n0 + wn + j * 16 + nlane;
      const float bv = bias[ncol];
#pragma unroll
      for (int r = 0; r < 8; ++r) {
        const int mrow = m0 + wm + i * 16 + mhalf + r;
        const float v = acc[i][j][r] + bv;
        if constexpr (MODE == 0) {
          ((float*)Out)[(size_t)mrow * N + ncol] = v;
        } else {
          const int b = mrow >> 11, sq = mrow & 2047;     // S = 2048
          const int h = ncol >> 6,  dh = ncol & 63;       // Dh = 64
          if constexpr (MODE == 1)
            ((__bf16*)Out)[(((size_t)(b * 16 + h)) * 2048 + sq) * 64 + dh] = (__bf16)v;
          else
            ((__bf16*)Out)[(((size_t)(b * 16 + h)) * 64 + dh) * 2048 + sq] = (__bf16)v;
        }
      }
    }
  }
}

// ===========================================================================
// Flash attention: per (b,h), 64 q-rows per block (4 waves x 16 rows),
// stream 32-key blocks: S = QK^T (2 accs, 4 WMMA), online softmax (f32,
// half-wave shfl_xor row reductions), P re-striped via per-wave LDS patch,
// PV accumulate (4 WMMA). scale = 1/sqrt(1024) = 1/32 folded into exp2.
// ===========================================================================
__global__ __launch_bounds__(128) void attention_wmma(
    const __bf16* __restrict__ Q,    // [B*H, S, Dh]
    const __bf16* __restrict__ Kt,   // [B*H, S, Dh]
    const __bf16* __restrict__ Vt,   // [B*H, Dh, S]
    __bf16* __restrict__ Ctx)        // [B, S, D] bf16
{
  constexpr int S = 2048, Dh = 64;
  constexpr float SCALE       = 0.03125f;          // 1/sqrt(1024)
  constexpr float SCALE_LOG2E = SCALE * LOG2E_F;

  const int lane = threadIdx.x & 31;
  const int wave = threadIdx.x >> 5;               // 0..3
  const int bh   = blockIdx.y;                     // 0..31
  const int q0   = blockIdx.x * 64 + wave * 16;

  __shared__ alignas(64) __bf16 Ps[4][16 * 32];    // per-wave P patch (1 KB)
  __bf16* pbuf = &Ps[wave][0];

  const __bf16* qbase = Q  + (size_t)bh * S * Dh;
  const __bf16* kbase = Kt + (size_t)bh * S * Dh;
  const __bf16* vbase = Vt + (size_t)bh * Dh * S;

  const int nlane  = lane & 15;
  const int b8     = (lane < 16) ? 0 : 16;
  const int koff16 = (lane < 16) ? 0 : 16;
  const int mhalf  = (lane >= 16) ? 8 : 0;

  // Q fragments (A-layout, K-dim = Dh = 64 -> two 16x32 frags), kept resident.
  v16bf a_q[2];
#pragma unroll
  for (int kk = 0; kk < 2; ++kk) {
    const __bf16* p = qbase + (size_t)(q0 + nlane) * Dh + kk * 32;
    const v8bf lo = *(const v8bf*)(p + b8);
    const v8bf hi = *(const v8bf*)(p + 8 + b8);
    v16bf a;
#pragma unroll
    for (int i = 0; i < 8; ++i) { a[i] = lo[i]; a[8 + i] = hi[i]; }
    a_q[kk] = a;
  }

  v8f ctx[4] = {};                                 // 16 x 64 f32 output tile
  float mrow[8], lrow[8];
#pragma unroll
  for (int r = 0; r < 8; ++r) { mrow[r] = -1e30f; lrow[r] = 0.0f; }

  for (int j = 0; j < S / 32; ++j) {
    // --- scores: S16x32 = Q(16x64) * K^T(64x32), two 16x16 accs ---
    v16bf bk[2][2];
#pragma unroll
    for (int g = 0; g < 2; ++g) {
      const __bf16* p = kbase + (size_t)(j * 32 + g * 16 + nlane) * Dh;
#pragma unroll
      for (int kk = 0; kk < 2; ++kk)
        bk[kk][g] = *(const v16bf*)(p + kk * 32 + koff16);
    }
    v8f sc[2];
#pragma unroll
    for (int g = 0; g < 2; ++g) {
      v8f z = {};
      z = WMMA_BF16(a_q[0], bk[0][g], z);
      z = WMMA_BF16(a_q[1], bk[1][g], z);
      sc[g] = z;
    }

    // --- online softmax: rows live in 16-lane halves; reduce across lanes ---
    float alpha[8];
#pragma unroll
    for (int r = 0; r < 8; ++r) {
      float v = fmaxf(sc[0][r], sc[1][r]);
      v = fmaxf(v, __shfl_xor(v, 1));
      v = fmaxf(v, __shfl_xor(v, 2));
      v = fmaxf(v, __shfl_xor(v, 4));
      v = fmaxf(v, __shfl_xor(v, 8));
      const float mnew = fmaxf(mrow[r], v * SCALE);
      alpha[r] = exp2f((mrow[r] - mnew) * LOG2E_F);
      mrow[r]  = mnew;
    }

#pragma unroll
    for (int r = 0; r < 8; ++r) {
      float rsum = 0.0f;
#pragma unroll
      for (int g = 0; g < 2; ++g) {
        const float p = exp2f(sc[g][r] * SCALE_LOG2E - mrow[r] * LOG2E_F);
        rsum += p;
        pbuf[(r + mhalf) * 32 + g * 16 + nlane] = (__bf16)p;  // re-stripe via LDS
      }
      rsum += __shfl_xor(rsum, 1);
      rsum += __shfl_xor(rsum, 2);
      rsum += __shfl_xor(rsum, 4);
      rsum += __shfl_xor(rsum, 8);
      lrow[r] = lrow[r] * alpha[r] + rsum;
    }
#pragma unroll
    for (int d = 0; d < 4; ++d)
#pragma unroll
      for (int r = 0; r < 8; ++r)
        ctx[d][r] *= alpha[r];

    // --- P as A-fragment (wave-private LDS, DS ops are in-order per wave) ---
    v16bf a_p;
    {
      const __bf16* p = pbuf + nlane * 32;
      const v8bf lo = *(const v8bf*)(p + b8);
      const v8bf hi = *(const v8bf*)(p + 8 + b8);
#pragma unroll
      for (int i = 0; i < 8; ++i) { a_p[i] = lo[i]; a_p[8 + i] = hi[i]; }
    }

    // --- PV: ctx(16x64) += P(16x32) * V(32x64); V^T rows give B-frag loads ---
#pragma unroll
    for (int d = 0; d < 4; ++d) {
      const int dh = d * 16 + nlane;
      const v16bf bv = *(const v16bf*)(vbase + (size_t)dh * S + j * 32 + koff16);
      ctx[d] = WMMA_BF16(a_p, bv, ctx[d]);
    }
  }

  // --- epilogue: ctx /= l, write bf16 ctx in [B,S,D] for output projection ---
  const int b = bh >> 4, h = bh & 15;
#pragma unroll
  for (int r = 0; r < 8; ++r) {
    const float inv = 1.0f / lrow[r];
    const int srow = q0 + mhalf + r;
    const size_t base = ((size_t)b * 2048 + srow) * 1024 + h * 64;
#pragma unroll
    for (int d = 0; d < 4; ++d)
      Ctx[base + d * 16 + nlane] = (__bf16)(ctx[d][r] * inv);
  }
}

// ===========================================================================
extern "C" void kernel_launch(void* const* d_in, const int* in_sizes, int n_in,
                              void* d_out, int out_size, void* d_ws, size_t ws_size,
                              hipStream_t stream)
{
  const float* X  = (const float*)d_in[0];
  const float* Wq = (const float*)d_in[1];
  const float* bq = (const float*)d_in[2];
  const float* Wk = (const float*)d_in[3];
  const float* bk = (const float*)d_in[4];
  const float* Wv = (const float*)d_in[5];
  const float* bv = (const float*)d_in[6];
  const float* Wo = (const float*)d_in[7];
  const float* bo = (const float*)d_in[8];

  // Workspace: 4 bf16 buffers of 4096x1024 = 8 MB each (32 MB total).
  const size_t MN = (size_t)4096 * 1024;
  char* wsb = (char*)d_ws;
  __bf16* q_ws  = (__bf16*)(wsb);
  __bf16* k_ws  = (__bf16*)(wsb + 2 * MN);
  __bf16* vT_ws = (__bf16*)(wsb + 4 * MN);
  __bf16* c_ws  = (__bf16*)(wsb + 6 * MN);

  dim3 gg(1024 / 128, 4096 / 128);   // (8, 32) blocks of 256 threads

  gemm_bias_wmma<0, 1><<<gg, 256, 0, stream>>>(X, Wq, bq, q_ws);
  gemm_bias_wmma<0, 1><<<gg, 256, 0, stream>>>(X, Wk, bk, k_ws);
  gemm_bias_wmma<0, 2><<<gg, 256, 0, stream>>>(X, Wv, bv, vT_ws);

  attention_wmma<<<dim3(2048 / 64, 32), 128, 0, stream>>>(q_ws, k_ws, vT_ws, c_ws);

  gemm_bias_wmma<1, 0><<<gg, 256, 0, stream>>>(c_ws, Wo, bo, (float*)d_out);
}
